// SelfAttention_23192823399168
// MI455X (gfx1250) — matrix operand order repaired and checked by reading, not compiled, not run on hardware
//
#include <hip/hip_runtime.h>
#include <hip/hip_bf16.h>

// Self-attention: B=4, N=2048, C=1024, fp32 in/out.  Compute-bound (~120 GFLOP
// vs ~200MB traffic -> ~8us at 23.3TB/s), so: 3-GEMM pipeline, all math on
// v_wmma_f32_16x16x32_f16 (fp32 accumulate), f16 intermediates in L2.
//   1) gemm_qkv:    Q,K (f16 row-major) and V^T (f16) from x,W,b (f32)
//   2) gemm_scores: S = Q K^T / 32  (f16, materialized)
//   3) softmax_rows: in-place row softmax on S
//   4) gemm_av:     out = P * V  (via V^T rows), f32
// GEMM shape: 256 thr = 8 wave32 (2M x 4N), WG tile 64x256, wave tile 32x64
// (2x4 WMMA accums), BK=32, double-buffered LDS.
// f16-source GEMMs stage via GLOBAL_LOAD_ASYNC_TO_LDS_B128 (ASYNCcnt),
// f32-source GEMM stages via VGPR cvt (needs v_cvt_pk_f16_f32 anyway).
// Workspace: Q 16MB | K 16MB | V^T 16MB | S 32MB = 80MB.

typedef __attribute__((ext_vector_type(16))) _Float16 v16h_t;
typedef __attribute__((ext_vector_type(8)))  _Float16 v8h_t;
typedef __attribute__((ext_vector_type(8)))  float    v8f_t;
typedef __attribute__((ext_vector_type(4)))  float    v4f_t;

union H16U { v16h_t v; v8h_t h[2]; };

#define LDS_STRIDE 40            // 32 halves + 8 pad (80B rows, 16B aligned)
#define A_ELEMS   (64  * LDS_STRIDE)
#define B_ELEMS   (256 * LDS_STRIDE)

// ---- CDNA5 async global->LDS copy (ISA 10.x / 15.18.3, ASYNCcnt) -----------
__device__ __forceinline__ void async_copy16(unsigned ldsAddr, const void* g) {
    asm volatile("global_load_async_to_lds_b128 %0, %1, off"
                 :: "v"(ldsAddr), "v"((unsigned long long)(uintptr_t)g)
                 : "memory");
}
__device__ __forceinline__ void wait_async0() {
    asm volatile("s_wait_asynccnt 0x0" ::: "memory");
}

// ---- WMMA fragment loaders (ISA 7.12.2 16-bit layouts, wave32) -------------
// A 16x32 (MxK): lane r(0..15) row=r; hi=0 -> K 0..7 & 16..23 ; hi=1 -> K 8..15 & 24..31
__device__ __forceinline__ v16h_t load_afrag(const _Float16* lA, int rowBase, int r, int hi) {
    H16U u;
    const _Float16* p = lA + (rowBase + r) * LDS_STRIDE;
    u.h[0] = *(const v8h_t*)(p + hi * 8);
    u.h[1] = *(const v8h_t*)(p + 16 + hi * 8);
    return u.v;
}
// B 32x16 (KxN): lane r = column; hi=0 -> K 0..7 & 16..23 ; hi=1 -> K 8..15 & 24..31
__device__ __forceinline__ v16h_t load_bfrag(const _Float16* lB, int rowBase, int r, int hi) {
    H16U u;
    const _Float16* p = lB + (rowBase + r) * LDS_STRIDE;
    u.h[0] = *(const v8h_t*)(p + hi * 8);
    u.h[1] = *(const v8h_t*)(p + 16 + hi * 8);
    return u.v;
}

#define WMMA_F16(A, B, C) \
    __builtin_amdgcn_wmma_f32_16x16x32_f16(false, (A), false, (B), (short)0, (C), false, false)

// 8 WMMAs per k-step: 2 A-frags x 4 B-frags
__device__ __forceinline__ void mma_step8(const _Float16* lA, const _Float16* lB,
                                          int wvM, int wvN, int r, int hi, v8f_t acc[2][4]) {
    v16h_t a[2], b[4];
    #pragma unroll
    for (int i = 0; i < 2; ++i) a[i] = load_afrag(lA, wvM * 32 + i * 16, r, hi);
    #pragma unroll
    for (int j = 0; j < 4; ++j) b[j] = load_bfrag(lB, wvN * 64 + j * 16, r, hi);
    #pragma unroll
    for (int i = 0; i < 2; ++i)
        #pragma unroll
        for (int j = 0; j < 4; ++j)
            acc[i][j] = WMMA_F16(a[i], b[j], acc[i][j]);
}

// ---- staging helpers -------------------------------------------------------
// A tile 64x32 halves: thread t -> row t>>2, chunk (t&3)*8 halves (16B)
// B tile 256x32 halves: thread t -> full row t (64B)
__device__ __forceinline__ void stA_cvt(_Float16* lA, int row, int ch, v4f_t f0, v4f_t f1) {
    v8h_t h;
    #pragma unroll
    for (int i = 0; i < 4; ++i) { h[i] = (_Float16)f0[i]; h[4 + i] = (_Float16)f1[i]; }
    *(v8h_t*)(lA + row * LDS_STRIDE + ch * 8) = h;
}
__device__ __forceinline__ void stB_cvt(_Float16* lB, int row, const v4f_t* f) {
    _Float16* p = lB + row * LDS_STRIDE;
    #pragma unroll
    for (int q = 0; q < 4; ++q) {
        v8h_t h;
        #pragma unroll
        for (int i = 0; i < 4; ++i) { h[i] = (_Float16)f[2*q][i]; h[4 + i] = (_Float16)f[2*q+1][i]; }
        *(v8h_t*)(p + q * 8) = h;
    }
}
// async (f16 source) variants
__device__ __forceinline__ void stA_async(_Float16* lA, const _Float16* gA, int row, int ch) {
    async_copy16((unsigned)(uintptr_t)(lA + row * LDS_STRIDE + ch * 8), gA);
}
__device__ __forceinline__ void stB_async(_Float16* lB, const _Float16* gB, int row) {
    const unsigned dst = (unsigned)(uintptr_t)(lB + row * LDS_STRIDE);
    #pragma unroll
    for (int q = 0; q < 4; ++q) async_copy16(dst + q * 16, gB + q * 8);
}

// ============================================================================
// Kernel 1: fused QKV projection.  y[n][d] = sum_c x[n][c]*W[d][c] + b[d]
// M=8192, N=3072 (Q|K|V), K=1024.  f32 sources converted during LDS staging.
// V third stored transposed: Vt[b][d][i] (contiguous per-lane stores).
// ============================================================================
__global__ void __launch_bounds__(256) gemm_qkv(
    const float* __restrict__ X,
    const float* __restrict__ Wq, const float* __restrict__ Wk, const float* __restrict__ Wv,
    const float* __restrict__ bq, const float* __restrict__ bk, const float* __restrict__ bv,
    _Float16* __restrict__ Qh, _Float16* __restrict__ Kh, _Float16* __restrict__ Vt)
{
    __shared__ alignas(16) _Float16 lA[2 * A_ELEMS];
    __shared__ alignas(16) _Float16 lB[2 * B_ELEMS];

    const int m0 = blockIdx.x * 64;
    const int n0 = blockIdx.y * 256;
    const int t  = threadIdx.x;
    const int wv = t >> 5, lane = t & 31, r = lane & 15, hi = lane >> 4;
    const int wvM = wv & 1, wvN = wv >> 1;
    const int arow = t >> 2, ach = t & 3;

    const int wsel = n0 >> 10;                       // scalar: Q / K / V block
    const float* Wbase = (wsel == 0 ? Wq : (wsel == 1 ? Wk : Wv));
    const float* Wrow  = Wbase + (size_t)((n0 + t) & 1023) * 1024;   // this thread's B row
    const float* Arow  = X + (size_t)(m0 + arow) * 1024 + ach * 8;   // this thread's A chunk

    v8f_t acc[2][4] = {};
    v4f_t af0, af1, bf[8];

    // prologue: stage k0 = 0
    af0 = *(const v4f_t*)(Arow); af1 = *(const v4f_t*)(Arow + 4);
    #pragma unroll
    for (int q = 0; q < 8; ++q) bf[q] = *(const v4f_t*)(Wrow + q * 4);
    stA_cvt(lA, arow, ach, af0, af1);
    stB_cvt(lB, t, bf);
    __syncthreads();

    int cur = 0;
    for (int k0 = 0; k0 < 1024; k0 += 32) {
        const int kn = k0 + 32;
        if (kn < 1024) {                             // issue next-tile loads
            af0 = *(const v4f_t*)(Arow + kn); af1 = *(const v4f_t*)(Arow + kn + 4);
            #pragma unroll
            for (int q = 0; q < 8; ++q) bf[q] = *(const v4f_t*)(Wrow + kn + q * 4);
            if (kn + 32 < 1024) {
                __builtin_prefetch(Arow + kn + 32, 0, 3);
                __builtin_prefetch(Wrow + kn + 32, 0, 3);
            }
        }
        mma_step8(lA + cur * A_ELEMS, lB + cur * B_ELEMS, wvM, wvN, r, hi, acc);
        if (kn < 1024) {
            stA_cvt(lA + (cur ^ 1) * A_ELEMS, arow, ach, af0, af1);
            stB_cvt(lB + (cur ^ 1) * B_ELEMS, t, bf);
        }
        __syncthreads();
        cur ^= 1;
    }

    // epilogue: bias add; Q/K row-major, V transposed
    const float* bsel = (wsel == 0 ? bq : (wsel == 1 ? bk : bv));
    #pragma unroll
    for (int j = 0; j < 4; ++j) {
        const int dc = n0 + wvN * 64 + j * 16 + r;
        const int cc = dc & 1023;
        const float bias = bsel[cc];
        #pragma unroll
        for (int i = 0; i < 2; ++i) {
            const int gmb = m0 + wvM * 32 + i * 16 + hi * 8;   // 8 consecutive rows
            if (wsel < 2) {
                _Float16* dst = (wsel == 0 ? Qh : Kh);
                #pragma unroll
                for (int v = 0; v < 8; ++v)
                    dst[(size_t)(gmb + v) * 1024 + cc] = (_Float16)(acc[i][j][v] + bias);
            } else {
                const int bb = gmb >> 11, is = gmb & 2047;
                v8h_t h;
                #pragma unroll
                for (int v = 0; v < 8; ++v) h[v] = (_Float16)(acc[i][j][v] + bias);
                *(v8h_t*)(Vt + ((size_t)bb * 1024 + cc) * 2048 + is) = h;
            }
        }
    }
}

// ============================================================================
// Kernel 2: S[b] = Q[b] K[b]^T * (1/32).  M=N=2048, K=1024 per batch.
// Staging via async global->LDS copies (no VGPR round-trip).
// ============================================================================
__global__ void __launch_bounds__(256) gemm_scores(
    const _Float16* __restrict__ Qh, const _Float16* __restrict__ Kh,
    _Float16* __restrict__ S)
{
    __shared__ alignas(16) _Float16 lA[2 * A_ELEMS];
    __shared__ alignas(16) _Float16 lB[2 * B_ELEMS];

    const int b  = blockIdx.z;
    const int m0 = blockIdx.x * 64;
    const int n0 = blockIdx.y * 256;
    const int t  = threadIdx.x;
    const int wv = t >> 5, lane = t & 31, r = lane & 15, hi = lane >> 4;
    const int wvM = wv & 1, wvN = wv >> 1;
    const int arow = t >> 2, ach = t & 3;

    const _Float16* Arow = Qh + (size_t)b * 2048 * 1024 + (size_t)(m0 + arow) * 1024 + ach * 8;
    const _Float16* Brow = Kh + (size_t)b * 2048 * 1024 + (size_t)(n0 + t) * 1024;
    _Float16* Sb = S + (size_t)b * 2048 * 2048;

    v8f_t acc[2][4] = {};
    stA_async(lA, Arow, arow, ach);
    stB_async(lB, Brow, t);
    wait_async0();
    __syncthreads();

    int cur = 0;
    for (int k0 = 0; k0 < 1024; k0 += 32) {
        const int kn = k0 + 32;
        if (kn < 1024) {                 // async-fill the other buffer
            stA_async(lA + (cur ^ 1) * A_ELEMS, Arow + kn, arow, ach);
            stB_async(lB + (cur ^ 1) * B_ELEMS, Brow + kn, t);
            if (kn + 32 < 1024) {
                __builtin_prefetch(Arow + kn + 32, 0, 3);
                __builtin_prefetch(Brow + kn + 32, 0, 3);
            }
        }
        mma_step8(lA + cur * A_ELEMS, lB + cur * B_ELEMS, wvM, wvN, r, hi, acc);
        if (kn < 1024) wait_async0();
        __syncthreads();
        cur ^= 1;
    }

    const float scale = 0.03125f;   // 1/sqrt(1024)
    #pragma unroll
    for (int j = 0; j < 4; ++j) {
        const int col = n0 + wvN * 64 + j * 16 + r;
        #pragma unroll
        for (int i = 0; i < 2; ++i) {
            const int gmb = m0 + wvM * 32 + i * 16 + hi * 8;
            #pragma unroll
            for (int v = 0; v < 8; ++v)
                Sb[(size_t)(gmb + v) * 2048 + col] = (_Float16)(acc[i][j][v] * scale);
        }
    }
}

// ============================================================================
// Kernel 3: in-place row softmax.  8192 rows x 2048 f16; f32 math.
// ============================================================================
__global__ void __launch_bounds__(256) softmax_rows(_Float16* __restrict__ S)
{
    __shared__ float red[256];
    const int t = threadIdx.x;
    _Float16* p = S + (size_t)blockIdx.x * 2048;

    v8h_t hv = *(const v8h_t*)(p + t * 8);
    float f[8], m = -1e30f;
    #pragma unroll
    for (int i = 0; i < 8; ++i) { f[i] = (float)hv[i]; m = fmaxf(m, f[i]); }

    red[t] = m; __syncthreads();
    #pragma unroll
    for (int s = 128; s > 0; s >>= 1) { if (t < s) red[t] = fmaxf(red[t], red[t + s]); __syncthreads(); }
    m = red[0]; __syncthreads();

    float e[8], sum = 0.f;
    #pragma unroll
    for (int i = 0; i < 8; ++i) { e[i] = __expf(f[i] - m); sum += e[i]; }
    red[t] = sum; __syncthreads();
    #pragma unroll
    for (int s = 128; s > 0; s >>= 1) { if (t < s) red[t] += red[t + s]; __syncthreads(); }
    const float inv = 1.0f / red[0];

    v8h_t o;
    #pragma unroll
    for (int i = 0; i < 8; ++i) o[i] = (_Float16)(e[i] * inv);
    *(v8h_t*)(p + t * 8) = o;
}

// ============================================================================
// Kernel 4: out[b] = P[b] * V[b]  (B-fragments read rows of V^T).
// M=2048, N=1024, K=2048 per batch.  f32 output.  Async LDS staging.
// ============================================================================
__global__ void __launch_bounds__(256) gemm_av(
    const _Float16* __restrict__ S, const _Float16* __restrict__ Vt,
    float* __restrict__ O)
{
    __shared__ alignas(16) _Float16 lA[2 * A_ELEMS];
    __shared__ alignas(16) _Float16 lB[2 * B_ELEMS];

    const int b  = blockIdx.z;
    const int m0 = blockIdx.x * 64;
    const int n0 = blockIdx.y * 256;
    const int t  = threadIdx.x;
    const int wv = t >> 5, lane = t & 31, r = lane & 15, hi = lane >> 4;
    const int wvM = wv & 1, wvN = wv >> 1;
    const int arow = t >> 2, ach = t & 3;

    const _Float16* Arow = S  + (size_t)b * 2048 * 2048 + (size_t)(m0 + arow) * 2048 + ach * 8;
    const _Float16* Brow = Vt + (size_t)b * 1024 * 2048 + (size_t)(n0 + t) * 2048;
    float* Ob = O + (size_t)b * 2048 * 1024;

    v8f_t acc[2][4] = {};
    stA_async(lA, Arow, arow, ach);
    stB_async(lB, Brow, t);
    wait_async0();
    __syncthreads();

    int cur = 0;
    for (int k0 = 0; k0 < 2048; k0 += 32) {
        const int kn = k0 + 32;
        if (kn < 2048) {
            stA_async(lA + (cur ^ 1) * A_ELEMS, Arow + kn, arow, ach);
            stB_async(lB + (cur ^ 1) * B_ELEMS, Brow + kn, t);
            if (kn + 32 < 2048) {
                __builtin_prefetch(Arow + kn + 32, 0, 3);
                __builtin_prefetch(Brow + kn + 32, 0, 3);
            }
        }
        mma_step8(lA + cur * A_ELEMS, lB + cur * B_ELEMS, wvM, wvN, r, hi, acc);
        if (kn < 2048) wait_async0();
        __syncthreads();
        cur ^= 1;
    }

    #pragma unroll
    for (int j = 0; j < 4; ++j) {
        const int col = n0 + wvN * 64 + j * 16 + r;
        #pragma unroll
        for (int i = 0; i < 2; ++i) {
            const int gmb = m0 + wvM * 32 + i * 16 + hi * 8;
            #pragma unroll
            for (int v = 0; v < 8; ++v)
                Ob[(size_t)(gmb + v) * 1024 + col] = acc[i][j][v];
        }
    }
}

// ============================================================================
extern "C" void kernel_launch(void* const* d_in, const int* in_sizes, int n_in,
                              void* d_out, int out_size, void* d_ws, size_t ws_size,
                              hipStream_t stream) {
    const float* x  = (const float*)d_in[0];
    const float* Wq = (const float*)d_in[1];
    const float* bq = (const float*)d_in[2];
    const float* Wk = (const float*)d_in[3];
    const float* bk = (const float*)d_in[4];
    const float* Wv = (const float*)d_in[5];
    const float* bv = (const float*)d_in[6];
    float* out = (float*)d_out;

    char* ws = (char*)d_ws;                                  // needs 80 MB
    _Float16* Qh = (_Float16*)(ws);                          // 16 MB
    _Float16* Kh = (_Float16*)(ws + ((size_t)16 << 20));     // 16 MB
    _Float16* Vt = (_Float16*)(ws + ((size_t)32 << 20));     // 16 MB
    _Float16* Sb = (_Float16*)(ws + ((size_t)48 << 20));     // 32 MB

    gemm_qkv    <<<dim3(128, 12, 1), 256, 0, stream>>>(x, Wq, Wk, Wv, bq, bk, bv, Qh, Kh, Vt);
    gemm_scores <<<dim3(32, 8, 4),   256, 0, stream>>>(Qh, Kh, Sb);
    softmax_rows<<<dim3(8192, 1, 1), 256, 0, stream>>>(Sb);
    gemm_av     <<<dim3(32, 4, 4),   256, 0, stream>>>(Sb, Vt, out);
}
